// Aggregation_Module_60894046323230
// MI455X (gfx1250) — matrix operand order repaired and checked by reading, not compiled, not run on hardware
//
#include <hip/hip_runtime.h>

// ---------------------------------------------------------------------------
// MI455X (gfx1250, wave32) implementation of the per-node-type HGT/GRU update.
//
// Roofline: 0.67 GFLOP vs 12.5 MB of HBM traffic -> memory bound (~0.5us at
// 23.3 TB/s) provided matmuls run on WMMA.  We bucket nodes by type (T=8) so
// 16-row WMMA tiles share one weight matrix, stage bf16-converted weights in
// LDS (130KB of the 320KB/WGP), and run all three einsums on
// v_wmma_f32_16x16x32_bf16 with f32 accumulation.
// ---------------------------------------------------------------------------

typedef __attribute__((ext_vector_type(16))) __bf16 v16bf;
typedef __attribute__((ext_vector_type(8)))  float  v8f;

#define NN    8192
#define TT    8
#define HIN   256     // H*IN
#define OUTF  64
#define G3    192     // 3*OUT
#define ROWS  64      // nodes per block
#define MAXTILES (NN / ROWS + TT)   // 136: upper bound on sum ceil(cnt[t]/64)

// workspace int layout
#define WS_COUNT 0
#define WS_OFF   8
#define WS_CUR   16
#define WS_TILE  24   // 9 entries (tileBase[0..7], total)
#define WS_PERM  40

// LDS row strides (in ushorts), padded to kill bank conflicts
#define RSX 258       // rows of 256 (stride 129 dwords -> +1 bank per row)
#define RSW 66        // rows of 64  (stride 33 dwords  -> odd, distinct banks)

union FragU { unsigned int u[8]; v16bf v; };

__device__ __forceinline__ unsigned short f2bf(float f) {
  unsigned int x = __float_as_uint(f);
  unsigned int r = x + 0x7FFFu + ((x >> 16) & 1u);   // round-to-nearest-even
  return (unsigned short)(r >> 16);
}

// A-matrix 16x32 bf16 fragment: lane holds row m=lane&15;
// k = (v/4)*16 + half*8 + (v&3)*2   (ISA 7.12.2, 16-bit A layout)
__device__ __forceinline__ v16bf fragA(const unsigned short* row, int kbase, int half) {
  FragU f;
#pragma unroll
  for (int v = 0; v < 8; ++v) {
    int k = kbase + ((v >> 2) << 4) + (half << 3) + ((v & 3) << 1);
    f.u[v] = *(const unsigned int*)(row + k);        // pair (k, k+1)
  }
  return f.v;
}

// B-matrix 32x16 bf16 fragment: lane holds column n=lane&15;
// k = (v/4)*16 + half*8 + (v&3)*2   (same k order as fragA)
__device__ __forceinline__ v16bf fragB(const unsigned short* col, int kbase, int half) {
  FragU f;
#pragma unroll
  for (int v = 0; v < 8; ++v) {
    int k = kbase + ((v >> 2) << 4) + (half << 3) + ((v & 3) << 1);
    f.u[v] = *(const unsigned int*)(col + k);
  }
  return f.v;
}

__device__ __forceinline__ v8f wmma(v16bf a, v16bf b, v8f c) {
  return __builtin_amdgcn_wmma_f32_16x16x32_bf16(false, a, false, b,
                                                 (short)0, c, false, false);
}

// ---------------- prep kernels: bucket nodes by type ----------------

__global__ void k_zero(int* ws) {
  int i = threadIdx.x;
  if (i < 40) ws[i] = 0;
}

__global__ void k_count(const int* __restrict__ nt, int* ws) {
  int n = blockIdx.x * blockDim.x + threadIdx.x;
  if (n < NN) atomicAdd(&ws[WS_COUNT + nt[n]], 1);
}

__global__ void k_scan(int* ws) {
  if (threadIdx.x == 0) {
    int off = 0, tiles = 0;
    for (int t = 0; t < TT; ++t) {
      int c = ws[WS_COUNT + t];
      ws[WS_OFF + t]  = off;
      ws[WS_TILE + t] = tiles;
      off   += c;
      tiles += (c + ROWS - 1) / ROWS;
    }
    ws[WS_TILE + TT] = tiles;
  }
}

__global__ void k_scatter(const int* __restrict__ nt, int* ws) {
  int n = blockIdx.x * blockDim.x + threadIdx.x;
  if (n < NN) {
    int t = nt[n];
    int pos = ws[WS_OFF + t] + atomicAdd(&ws[WS_CUR + t], 1);
    ws[WS_PERM + pos] = n;
  }
}

// ---------------- main fused kernel: 4 waves, 64 nodes / block ----------------

__global__ __launch_bounds__(128, 1)
void hgru_main(const float* __restrict__ agg_msg,
               const float* __restrict__ hmat,
               const int*   __restrict__ ws_i,
               const float* __restrict__ W_att, const float* __restrict__ b_att,
               const float* __restrict__ W_in,  const float* __restrict__ W_hid,
               const float* __restrict__ b_in,  const float* __restrict__ b_hid,
               float* __restrict__ out)
{
  __shared__ unsigned short s_wa[OUTF * RSX];   // W_att^T  [n][k] 33KB
  __shared__ unsigned short s_x [ROWS * RSX];   // relu(x)  [row][k] 33KB
  __shared__ unsigned short s_wi[G3 * RSW];     // W_in^T   [n][k] 25KB
  __shared__ unsigned short s_wh[G3 * RSW];     // W_hid^T  [n][k] 25KB
  __shared__ unsigned short s_y [ROWS * RSW];   // stage-1 out 8.4KB
  __shared__ unsigned short s_hh[ROWS * RSW];   // h (bf16)    8.4KB
  __shared__ int s_nid[ROWS];

  int b = blockIdx.x;
  if (b >= ws_i[WS_TILE + TT]) return;
  int t = 0;
  while (t < TT - 1 && b >= ws_i[WS_TILE + t + 1]) ++t;
  int typeOff = ws_i[WS_OFF + t];
  int typeCnt = ws_i[WS_COUNT + t];
  int posBase = typeOff + (b - ws_i[WS_TILE + t]) * ROWS;
  const int* perm = ws_i + WS_PERM;

  int tid = threadIdx.x;

  // ---- stage node ids + bf16 weights into LDS ----
  if (tid < ROWS) {
    int p = posBase + tid;
    s_nid[tid] = (p < typeOff + typeCnt) ? perm[p] : -1;
  }
  const float* wa = W_att + t * (HIN * OUTF);
  for (int e = tid; e < HIN * OUTF; e += 128) {      // coalesced read, transpose
    int k = e >> 6, n = e & 63;
    s_wa[n * RSX + k] = f2bf(wa[e]);
  }
  const float* wi = W_in  + t * (OUTF * G3);
  const float* wh = W_hid + t * (OUTF * G3);
  for (int e = tid; e < OUTF * G3; e += 128) {
    int k = e / G3, n = e - k * G3;
    s_wi[n * RSW + k] = f2bf(wi[e]);
    s_wh[n * RSW + k] = f2bf(wh[e]);
  }
  __syncthreads();                                    // s_nid ready

  // ---- stage relu(agg_msg) and h (gathered rows) ----
  for (int e = tid; e < ROWS * HIN; e += 128) {
    int r = e >> 8, c = e & 255;
    int nid = s_nid[r];
    float v = (nid >= 0) ? agg_msg[nid * HIN + c] : 0.0f;
    s_x[r * RSX + c] = f2bf(fmaxf(v, 0.0f));
  }
  for (int e = tid; e < ROWS * OUTF; e += 128) {
    int r = e >> 6, c = e & 63;
    int nid = s_nid[r];
    float v = (nid >= 0) ? hmat[nid * OUTF + c] : 0.0f;
    s_hh[r * RSW + c] = f2bf(v);
  }
  __syncthreads();

  int lane = tid & 31;
  int wave = tid >> 5;
  int half = lane >> 4;
  int nn   = lane & 15;
  int arow = wave * 16 + nn;                 // this lane's A row / D column id
  const v8f vzero = {0.f, 0.f, 0.f, 0.f, 0.f, 0.f, 0.f, 0.f};

  // ---- stage 1: y = relu(x @ W_att + b_att), 16x256 @ 256x64 ----
  const unsigned short* xrow = s_x + arow * RSX;
  v8f accY[4];
#pragma unroll
  for (int q = 0; q < 4; ++q) accY[q] = vzero;
  for (int kk = 0; kk < HIN; kk += 32) {
    v16bf a = fragA(xrow, kk, half);
#pragma unroll
    for (int q = 0; q < 4; ++q) {
      v16bf bm = fragB(s_wa + (q * 16 + nn) * RSX, kk, half);
      accY[q] = wmma(a, bm, accY[q]);
    }
  }
  const float* bav = b_att + t * OUTF;
#pragma unroll
  for (int q = 0; q < 4; ++q) {
    int c = q * 16 + nn;
    float bv = bav[c];
#pragma unroll
    for (int j = 0; j < 8; ++j) {
      int m = j + half * 8;                  // D layout: VGPR j -> M=j / j+8
      float y = fmaxf(accY[q][j] + bv, 0.0f);
      s_y[(wave * 16 + m) * RSW + c] = f2bf(y);   // own rows only: no barrier
    }
  }

  const unsigned short* yrow = s_y  + arow * RSW;
  const unsigned short* hrow = s_hh + arow * RSW;

  // ---- stage 2: fused r/z gates: rz = y@W_in[:,:128] + h@W_hid[:,:128] ----
  v8f accRZ[8];
#pragma unroll
  for (int q = 0; q < 8; ++q) accRZ[q] = vzero;
#pragma unroll
  for (int kk = 0; kk < 2; ++kk) {
    v16bf ay = fragA(yrow, kk * 32, half);
#pragma unroll
    for (int q = 0; q < 8; ++q) {
      v16bf bm = fragB(s_wi + (q * 16 + nn) * RSW, kk * 32, half);
      accRZ[q] = wmma(ay, bm, accRZ[q]);
    }
    v16bf ah = fragA(hrow, kk * 32, half);
#pragma unroll
    for (int q = 0; q < 8; ++q) {
      v16bf bm = fragB(s_wh + (q * 16 + nn) * RSW, kk * 32, half);
      accRZ[q] = wmma(ah, bm, accRZ[q]);
    }
  }

  // ---- stage 3: n-gate halves separately (mixed through r) ----
  v8f accIN[4], accHN[4];
#pragma unroll
  for (int q = 0; q < 4; ++q) { accIN[q] = vzero; accHN[q] = vzero; }
#pragma unroll
  for (int kk = 0; kk < 2; ++kk) {
    v16bf ay = fragA(yrow, kk * 32, half);
    v16bf ah = fragA(hrow, kk * 32, half);
#pragma unroll
    for (int q = 0; q < 4; ++q) {
      int c = 128 + q * 16 + nn;
      v16bf bi2 = fragB(s_wi + c * RSW, kk * 32, half);
      v16bf bh2 = fragB(s_wh + c * RSW, kk * 32, half);
      accIN[q] = wmma(ay, bi2, accIN[q]);
      accHN[q] = wmma(ah, bh2, accHN[q]);
    }
  }

  // ---- GRU gating (f32 VALU) + scatter ----
  const float* biv = b_in  + t * G3;
  const float* bhv = b_hid + t * G3;
#pragma unroll
  for (int q = 0; q < 4; ++q) {
    int c = q * 16 + nn;
    float br  = biv[c]       + bhv[c];
    float bz  = biv[c + 64]  + bhv[c + 64];
    float bni = biv[c + 128];
    float bnh = bhv[c + 128];
#pragma unroll
    for (int j = 0; j < 8; ++j) {
      int m = j + half * 8;
      int nid = s_nid[wave * 16 + m];
      if (nid < 0) continue;
      float r  = 1.0f / (1.0f + __expf(-(accRZ[q][j] + br)));
      float z  = 1.0f / (1.0f + __expf(-(accRZ[q + 4][j] + bz)));
      float ng = tanhf((accIN[q][j] + bni) + r * (accHN[q][j] + bnh));
      float hv = hmat[nid * OUTF + c];
      out[nid * OUTF + c] = (1.0f - z) * ng + z * hv;
    }
  }
}

// ---------------------------------------------------------------------------

extern "C" void kernel_launch(void* const* d_in, const int* in_sizes, int n_in,
                              void* d_out, int out_size, void* d_ws, size_t ws_size,
                              hipStream_t stream) {
  const float* agg   = (const float*)d_in[0];
  const float* hmat  = (const float*)d_in[1];
  const int*   nt    = (const int*)  d_in[2];
  const float* W_att = (const float*)d_in[3];
  const float* b_att = (const float*)d_in[4];
  const float* W_in  = (const float*)d_in[5];
  const float* W_hid = (const float*)d_in[6];
  const float* b_in  = (const float*)d_in[7];
  const float* b_hid = (const float*)d_in[8];
  float* out = (float*)d_out;
  int*   ws  = (int*)d_ws;

  k_zero   <<<1,        64,  0, stream>>>(ws);
  k_count  <<<NN / 256, 256, 0, stream>>>(nt, ws);
  k_scan   <<<1,        32,  0, stream>>>(ws);
  k_scatter<<<NN / 256, 256, 0, stream>>>(nt, ws);
  hgru_main<<<MAXTILES, 128, 0, stream>>>(agg, hmat, ws, W_att, b_att,
                                          W_in, W_hid, b_in, b_hid, out);
}
